// DeconvContentDecoder_6536940224699
// MI455X (gfx1250) — matrix operand order repaired and checked by reading, not compiled, not run on hardware
//
#include <hip/hip_runtime.h>
#include <hip/hip_bf16.h>
#include <stdint.h>

// ---------------------------------------------------------------------------
// DeconvContentDecoder for MI455X (gfx1250, wave32, WMMA, async-LDS).
//
//   dec0:   h[16][128]                 = lrelu(x @ dec0_w^T + b)
//   stage1..4: stride-2 K=4 convT as 8 parity sub-convs, each a GEMM
//     M = 16*n^3 (batch x base voxel), K = 8*cin (ci, 2x2x2 tap), N = cout.
//   One WG per (part, parity): full KxN f16 weight panel staged in LDS once,
//   then M tiles streamed: im2col A-block gathered as b32 *tap pairs*
//   (w-halo-padded activations + zero page -> branchless, 2 halves/load,
//   via GLOBAL_LOAD_ASYNC_TO_LDS_B32), then v_wmma_f32_16x16x32_f16.
//
// K-order per dim (parity bit pd, tap bit td, base voxel coord v):
//   i = v - 1 + pd + td   (memory-increasing => tap pairs contiguous in w)
//   k = (pd ? 2 : 3) - 2*td
// Roofline: 474 MB f32 weights streamed once ~= 20us @ 23.3 TB/s dominates;
// parity/part re-reads of ~MB-scale per-part working sets hit the 192MB L2.
// ---------------------------------------------------------------------------

typedef _Float16 v16h __attribute__((ext_vector_type(16)));
typedef _Float16 v8h  __attribute__((ext_vector_type(8)));
typedef float    v8f  __attribute__((ext_vector_type(8)));
typedef uint32_t una_u32 __attribute__((aligned(2)));

#define LRELU_SLOPE 0.2f
#define NPARTS 64
#define NB     16

#if __has_builtin(__builtin_amdgcn_global_load_async_to_lds_b32) && \
    __has_builtin(__builtin_amdgcn_s_wait_asynccnt)
#define ASYNC_LDS 1
typedef __attribute__((address_space(1))) int g_as_i32;
typedef __attribute__((address_space(3))) int l_as_i32;
#endif

static __device__ __forceinline__ v8f wmma_f16f32(v16h a, v16h b, v8f c) {
  // (neg_a, A, neg_b, B, c_mod, C, reuse_a, reuse_b)
  return __builtin_amdgcn_wmma_f32_16x16x32_f16(false, a, false, b, (short)0, c,
                                                false, false);
}

// Copy one f16 tap-pair (4B) global -> LDS (LDS side is 4B aligned).
static __device__ __forceinline__ void pair_copy(const _Float16* g,
                                                 _Float16* l) {
#ifdef ASYNC_LDS
  __builtin_amdgcn_global_load_async_to_lds_b32(
      (g_as_i32*)(unsigned long)(const void*)g,
      (l_as_i32*)(unsigned int)(unsigned long)(void*)l, 0, 0);
#else
  *(uint32_t*)l = *(const una_u32*)g;
#endif
}

static __device__ __forceinline__ void gather_fence() {
#ifdef ASYNC_LDS
  __builtin_amdgcn_s_wait_asynccnt(0);
#endif
  __syncthreads();
}

// ---------------------------------------------------------------------------
// dec0: h = lrelu(x @ dec0_w^T + bias). M=16,N=128,K=256. One WG, 8 waves.
// Output into w-halo layout h16p[b][c][3] at index 1 (halo pre-zeroed).
// ---------------------------------------------------------------------------
__global__ __launch_bounds__(256)
void dec0_kernel(const float* __restrict__ x, const float* __restrict__ w,
                 const float* __restrict__ bias, _Float16* __restrict__ h16p) {
  const int lane = threadIdx.x & 31;
  const int wave = threadIdx.x >> 5;
  const int nl   = lane & 15;
  const int hib  = lane >> 4;
  const int hi8  = hib << 3;
  const int c    = wave * 16 + nl;
  v8f acc = {};
  for (int kk0 = 0; kk0 < 256; kk0 += 32) {
    v16h a, bf;
#pragma unroll
    for (int j = 0; j < 16; ++j) {
      int ka = kk0 + ((j < 8) ? j : j + 8) + hi8;  // A frag striping
      int kb = kk0 + ((j < 8) ? j : j + 8) + hi8;  // B frag striping
      a[j]  = (_Float16)x[nl * 256 + ka];
      bf[j] = (_Float16)w[c * 256 + kb];
    }
    acc = wmma_f16f32(a, bf, acc);
  }
#pragma unroll
  for (int r = 0; r < 8; ++r) {
    int b = r + hi8;
    float v = acc[r] + bias[c];
    v = v > 0.0f ? v : v * LRELU_SLOPE;
    h16p[(b * 128 + c) * 3 + 1] = (_Float16)v;
  }
}

// ---------------------------------------------------------------------------
// Generic convT parity-GEMM stage (compile-time shapes). grid.x = NPARTS*8,
// block = 256 (8 waves tiled MT x NT, MT*NT == 8). M % MBLK == 0 by choice.
// ---------------------------------------------------------------------------
template <int LN, int CIN, int COUT, int MT, int NT, int IN_PART, int LRELU,
          int FINAL>
__global__ __launch_bounds__(256)
void convt_stage(const _Float16* __restrict__ in, const float* __restrict__ w,
                 const float* __restrict__ bias, _Float16* __restrict__ out16,
                 float* __restrict__ out32, const _Float16* __restrict__ zpage) {
  constexpr int N = 1 << LN, N3 = 1 << (3 * LN);
  constexpr int M = NB * N3;
  constexpr int ON = 2 * N, ON3 = ON * ON * ON;
  constexpr int WP_IN = N + 2, WP_OUT = ON + 2;   // w-halo pitches
  constexpr int KTOT = CIN * 8;
  constexpr int NBLK = NT * 16, MBLK = MT * 16;
  constexpr int WPITCH = KTOT + 8;                // 16B skew
  constexpr int APITCH = 72;                      // 64 halves + 16B skew
  constexpr int PER_T = MBLK * 32 / 256;          // tap-pairs per thread/kstep

  extern __shared__ _Float16 smem[];
  _Float16* Wl = smem;                            // [NBLK][WPITCH]
  _Float16* Al = smem + (size_t)NBLK * WPITCH;    // [MBLK][APITCH]

  const int q = blockIdx.x & 7;    // parity (inner => per-part L2 reuse)
  const int p = blockIdx.x >> 3;   // part
  const int pd = (q >> 2) & 1, ph = (q >> 1) & 1, pw = q & 1;

  const int tid  = threadIdx.x;
  const int lane = tid & 31;
  const int wave = tid >> 5;
  const int mt   = wave % MT;
  const int nt   = wave / MT;
  const int hib  = lane >> 4;
  const int hi8  = hib << 3;
  const int nl   = lane & 15;

  // ---- stage full KxN weight panel (f32 -> f16) for (p,q) into LDS ----
  for (int e = tid; e < KTOT * NBLK; e += 256) {
    int col = e / KTOT;
    int kk  = e - col * KTOT;
    int ci  = kk >> 3, tap = kk & 7;
    int td = (tap >> 2) & 1, th = (tap >> 1) & 1, tw = tap & 1;
    int kd = (pd ? 2 : 3) - 2 * td;
    int kh = (ph ? 2 : 3) - 2 * th;
    int kw = (pw ? 2 : 3) - 2 * tw;
    float wv = 0.0f;
    if (col < COUT)
      wv = w[(((p * CIN + ci) * COUT + col) << 6) + kd * 16 + kh * 4 + kw];
    Wl[col * WPITCH + kk] = (_Float16)wv;
  }
  __syncthreads();

  const int pbase = (IN_PART ? p : 0) * NB;

  for (int m0 = 0; m0 < M; m0 += MBLK) {
    v8f acc = {};
    for (int kk0 = 0; kk0 < KTOT; kk0 += 64) {
      __syncthreads();  // all waves finished reading previous A block
      // ---- branchless im2col gather: MBLK x 64 halves as b32 tap pairs ----
#pragma unroll
      for (int i = 0; i < PER_T; ++i) {
        int e    = tid + i * 256;
        int mlcl = e >> 5;
        int pj   = e & 31;
        int ci   = (kk0 >> 3) + (pj >> 2);
        int td   = (pj >> 1) & 1, th = pj & 1;
        int mg = m0 + mlcl;
        int b  = mg >> (3 * LN);
        int v  = mg & (N3 - 1);
        int vd = v >> (2 * LN), vh = (v >> LN) & (N - 1), vw = v & (N - 1);
        int id = vd - 1 + pd + td;
        int ih = vh - 1 + ph + th;
        bool ok = ((unsigned)id < (unsigned)N) & ((unsigned)ih < (unsigned)N);
        int base = (((pbase + b) * CIN + ci) * N + id) * N + ih;
        const _Float16* g = ok ? (in + base * WP_IN + (vw + pw)) : zpage;
        pair_copy(g, &Al[mlcl * APITCH + (pj << 1)]);
      }
      gather_fence();
      // ---- fragments + 2x WMMA ----
      const _Float16* ap = &Al[(mt * 16 + nl) * APITCH];
      const _Float16* bp = &Wl[(nt * 16 + nl) * WPITCH + kk0];
#pragma unroll
      for (int half = 0; half < 2; ++half) {
        v8h alo = *(const v8h*)(ap + half * 32 + hi8);
        v8h ahi = *(const v8h*)(ap + half * 32 + 16 + hi8);
        v8h blo = *(const v8h*)(bp + half * 32 + hi8);
        v8h bhi = *(const v8h*)(bp + half * 32 + 16 + hi8);
        v16h a, bf;
#pragma unroll
        for (int j = 0; j < 8; ++j) {
          a[j] = alo[j]; a[j + 8] = ahi[j];    // K = (j<8?j:j+8)+hi8
          bf[j] = blo[j]; bf[j + 8] = bhi[j];  // K = (j<8?j:j+8)+hi8
        }
        acc = wmma_f16f32(a, bf, acc);
      }
    }
    // ---- epilogue: bias (+lrelu), store ----
    const int co = nt * 16 + nl;
#pragma unroll
    for (int r = 0; r < 8; ++r) {
      int mg = m0 + mt * 16 + r + hi8;
      int b  = mg >> (3 * LN);
      int v  = mg & (N3 - 1);
      int vd = v >> (2 * LN), vh = (v >> LN) & (N - 1), vw = v & (N - 1);
      int od = 2 * vd + pd, oh = 2 * vh + ph, ow = 2 * vw + pw;
      if (co < COUT) {
        float val = acc[r] + bias[p * COUT + co];
        if (LRELU) val = val > 0.0f ? val : val * LRELU_SLOPE;
        if (FINAL) {  // dense f32 out: [b][P][co][16^3]
          int off =
              ((b * NPARTS + p) * COUT + co) * ON3 + (od * ON + oh) * ON + ow;
          __builtin_nontemporal_store(val, &out32[off]);
        } else {      // f16 act, w-halo: [p][B][co][ON][ON][ON+2]
          int off = (((((p * NB + b) * COUT + co) * ON + od) * ON + oh)) * WP_OUT +
                    (ow + 1);
          out16[off] = (_Float16)val;
        }
      }
    }
  }
}

// ---------------------------------------------------------------------------
extern "C" void kernel_launch(void* const* d_in, const int* in_sizes, int n_in,
                              void* d_out, int out_size, void* d_ws,
                              size_t ws_size, hipStream_t stream) {
  const float* x      = (const float*)d_in[0];
  const float* dec0_w = (const float*)d_in[1];
  const float* dec0_b = (const float*)d_in[2];
  const float* w1 = (const float*)d_in[3];
  const float* b1 = (const float*)d_in[4];
  const float* w2 = (const float*)d_in[5];
  const float* b2 = (const float*)d_in[6];
  const float* w3 = (const float*)d_in[7];
  const float* b3 = (const float*)d_in[8];
  const float* w4 = (const float*)d_in[9];
  const float* b4 = (const float*)d_in[10];
  float* out = (float*)d_out;

  // Workspace: zero page + w-halo-padded f16 activations (~101 MB).
  _Float16* zp  = (_Float16*)d_ws;                          // 128 halves
  _Float16* h16 = zp + 128;                                 // 16*128*3
  _Float16* y1  = h16 + (size_t)NB * 128 * 3;               // 64*16*128*2*2*4
  _Float16* y2  = y1 + (size_t)NPARTS * NB * 128 * 2 * 2 * 4;
  _Float16* y3  = y2 + (size_t)NPARTS * NB * 64 * 4 * 4 * 6;
  size_t need = (128 + (size_t)NB * 128 * 3 +
                 (size_t)NPARTS * NB * 128 * 2 * 2 * 4 +
                 (size_t)NPARTS * NB * 64 * 4 * 4 * 6 +
                 (size_t)NPARTS * NB * 64 * 8 * 8 * 10) * sizeof(_Float16);
  if (ws_size < need) return;

  (void)hipMemsetAsync(d_ws, 0, need, stream);  // zero page + halos

  dec0_kernel<<<1, 256, 0, stream>>>(x, dec0_w, dec0_b, h16);

  // smem bytes = (NBLK*(KTOT+8) + MBLK*72) * 2
  size_t sm1 = ((size_t)128 * (1024 + 8) + 16 * 72) * sizeof(_Float16);
  convt_stage<0, 128, 128, 1, 8, 0, 1, 0>
      <<<NPARTS * 8, 256, sm1, stream>>>(h16, w1, b1, y1, nullptr, zp);

  size_t sm2 = ((size_t)64 * (1024 + 8) + 32 * 72) * sizeof(_Float16);
  convt_stage<1, 128, 64, 2, 4, 1, 1, 0>
      <<<NPARTS * 8, 256, sm2, stream>>>(y1, w2, b2, y2, nullptr, zp);

  size_t sm3 = ((size_t)64 * (512 + 8) + 32 * 72) * sizeof(_Float16);
  convt_stage<2, 64, 64, 2, 4, 1, 1, 0>
      <<<NPARTS * 8, 256, sm3, stream>>>(y2, w3, b3, y3, nullptr, zp);

  size_t sm4 = ((size_t)16 * (512 + 8) + 128 * 72) * sizeof(_Float16);
  convt_stage<3, 64, 4, 8, 1, 1, 0, 1>
      <<<NPARTS * 8, 256, sm4, stream>>>(y3, w4, b4, nullptr, out, zp);
}
